// LoRAMLPWrapper_26250840113719
// MI455X (gfx1250) — compile-verified, matrix operand order checked by reading, not run on hardware
//
#include <hip/hip_runtime.h>
#include <hip/hip_bf16.h>

// ---------------------------------------------------------------------------
// CDNA5 (gfx1250) LoRA-MLP (SwiGLU)
//   - one-shot fp32->bf16 conversion prepass (weights + x)
//   - double-buffered GLOBAL_LOAD_ASYNC_TO_LDS_B128 staging (ASYNCcnt)
//   - v_wmma_f32_16x16x32_bf16 main loops, fp32 accumulate
// ---------------------------------------------------------------------------

typedef __attribute__((ext_vector_type(16))) __bf16 v16bf;
typedef __attribute__((ext_vector_type(8)))  float  v8f;

#define TM 128   // block tile M
#define TN 64    // block tile N
#define TK 32    // K step (bf16 WMMA K)

// ---- fp32 <-> bf16 helpers (round-to-nearest-even) ----
__device__ __forceinline__ unsigned int pk2bf(float lo, float hi) {
    union { float f; unsigned u; } a, b;
    a.f = lo; b.f = hi;
    unsigned ua = a.u + (0x7FFFu + ((a.u >> 16) & 1u));
    unsigned ub = b.u + (0x7FFFu + ((b.u >> 16) & 1u));
    return (ua >> 16) | (ub & 0xFFFF0000u);
}
__device__ __forceinline__ unsigned short f2bf(float f) {
    union { float f; unsigned u; } v; v.f = f;
    unsigned u = v.u + (0x7FFFu + ((v.u >> 16) & 1u));
    return (unsigned short)(u >> 16);
}
__device__ __forceinline__ float bf2f(unsigned short h) {
    union { float f; unsigned u; } v; v.u = ((unsigned)h) << 16;
    return v.f;
}

// ---- CDNA5 async copy: global -> LDS, 16B per lane, tracked by ASYNCcnt ----
__device__ __forceinline__ void async_ld_b128(unsigned ldsOff, const void* g) {
    asm volatile("global_load_async_to_lds_b128 %0, %1, off"
                 :: "v"(ldsOff), "v"((unsigned long long)(size_t)g)
                 : "memory");
}
__device__ __forceinline__ void wait_async0() {
    asm volatile("s_wait_asynccnt 0x0" ::: "memory");
}
__device__ __forceinline__ unsigned lds_off(const void* p) {
    return (unsigned)(unsigned long long)(size_t)p;   // low 32 bits = LDS offset
}

// ---------------------------------------------------------------------------
// Elementwise fp32 -> bf16 conversion (one shot; amortized across all tile
// reuse in the GEMMs). n must be a multiple of 8.
// ---------------------------------------------------------------------------
__global__ __launch_bounds__(256)
void cvt_f32_bf16_kernel(const float* __restrict__ src,
                         unsigned short* __restrict__ dst,
                         long long n) {
    long long i = ((long long)blockIdx.x * 256 + threadIdx.x) * 8;
    if (i + 8 <= n) {
        float4 a = *(const float4*)(src + i);
        float4 b = *(const float4*)(src + i + 4);
        uint4 o;
        o.x = pk2bf(a.x, a.y); o.y = pk2bf(a.z, a.w);
        o.z = pk2bf(b.x, b.y); o.w = pk2bf(b.z, b.w);
        *(uint4*)(dst + i) = o;
    }
}

// ---------------------------------------------------------------------------
// xa = x @ a^T for two rank-16 adapters at once. One block per token row.
// ---------------------------------------------------------------------------
__global__ __launch_bounds__(256)
void xa_pair_f32_kernel(const float* __restrict__ x,
                        const float* __restrict__ a0,
                        const float* __restrict__ a1,
                        float* __restrict__ o0,
                        float* __restrict__ o1,
                        int K) {
    __shared__ float red0[256];
    __shared__ float red1[256];
    const int m   = blockIdx.x;
    const int t   = threadIdx.x;
    const int r   = t & 15;
    const int seg = t >> 4;
    const int L   = K >> 4;
    const float* xp = x  + (size_t)m * K + seg * L;
    const float* p0 = a0 + (size_t)r * K + seg * L;
    const float* p1 = a1 + (size_t)r * K + seg * L;
    float s0 = 0.f, s1 = 0.f;
    for (int i = 0; i < L; i += 4) {
        float4 xv = *(const float4*)(xp + i);
        float4 v0 = *(const float4*)(p0 + i);
        float4 v1 = *(const float4*)(p1 + i);
        s0 += xv.x * v0.x + xv.y * v0.y + xv.z * v0.z + xv.w * v0.w;
        s1 += xv.x * v1.x + xv.y * v1.y + xv.z * v1.z + xv.w * v1.w;
    }
    red0[t] = s0; red1[t] = s1;
    __syncthreads();
    if (t < 16) {
        float t0 = 0.f, t1 = 0.f;
        #pragma unroll
        for (int s = 0; s < 16; ++s) { t0 += red0[s * 16 + t]; t1 += red1[s * 16 + t]; }
        o0[m * 16 + t] = t0;
        o1[m * 16 + t] = t1;
    }
}

// Same, single adapter, bf16 activations (hidden @ down_a^T).
__global__ __launch_bounds__(256)
void xa_single_bf16_kernel(const unsigned short* __restrict__ h,
                           const float* __restrict__ a0,
                           float* __restrict__ o0,
                           int K) {
    __shared__ float red0[256];
    const int m   = blockIdx.x;
    const int t   = threadIdx.x;
    const int r   = t & 15;
    const int seg = t >> 4;
    const int L   = K >> 4;                // 688
    const unsigned short* hp = h + (size_t)m * K + seg * L;
    const float*          p0 = a0 + (size_t)r * K + seg * L;
    float s0 = 0.f;
    for (int i = 0; i < L; i += 4) {
        uint2  hv = *(const uint2*)(hp + i);
        float4 av = *(const float4*)(p0 + i);
        s0 += bf2f((unsigned short)(hv.x)) * av.x
            + bf2f((unsigned short)(hv.x >> 16)) * av.y
            + bf2f((unsigned short)(hv.y)) * av.z
            + bf2f((unsigned short)(hv.y >> 16)) * av.w;
    }
    red0[t] = s0;
    __syncthreads();
    if (t < 16) {
        float t0 = 0.f;
        #pragma unroll
        for (int s = 0; s < 16; ++s) t0 += red0[s * 16 + t];
        o0[m * 16 + t] = t0;
    }
}

// ---------------------------------------------------------------------------
// Fused gate/up GEMM (all-bf16 operands) + LoRA + SiLU*up -> hidden (bf16).
// Block tile 128x64, 8 wave32 waves each owning 32x32 (2x2 WMMA tiles x2
// matrices). Double-buffered async-DMA LDS staging overlapped with WMMA.
// ---------------------------------------------------------------------------
__global__ __launch_bounds__(256)
void gateup_wmma_kernel(const unsigned short* __restrict__ xb,   // [M,K] bf16
                        const unsigned short* __restrict__ gwb,  // [H,K] bf16
                        const unsigned short* __restrict__ uwb,  // [H,K] bf16
                        const float* __restrict__ gb,            // [H,16]
                        const float* __restrict__ ub,            // [H,16]
                        const float* __restrict__ xag,           // [M,16]
                        const float* __restrict__ xau,           // [M,16]
                        unsigned short* __restrict__ hidden,     // [M,H] bf16
                        int H, int K) {
    __shared__ __align__(32) unsigned short sA[2][TM * TK];
    __shared__ __align__(32) unsigned short sG[2][TN * TK];
    __shared__ __align__(32) unsigned short sU[2][TN * TK];

    const int t     = threadIdx.x;
    const int lane  = t & 31;
    const int wave  = t >> 5;
    const int waveM = (wave >> 1) * 32;    // 0,32,64,96
    const int waveN = (wave & 1) * 32;     // 0,32
    const int blockM = blockIdx.y * TM;
    const int blockN = blockIdx.x * TN;

    v8f accG[2][2], accU[2][2];
    #pragma unroll
    for (int i = 0; i < 2; ++i)
        #pragma unroll
        for (int j = 0; j < 2; ++j)
            #pragma unroll
            for (int e = 0; e < 8; ++e) { accG[i][j][e] = 0.f; accU[i][j][e] = 0.f; }

    // per-thread copy slots (bf16 tiles): A 32B/thread, G/U 16B/thread
    const int arow = t >> 1;            // 0..127
    const int acol = (t & 1) * 16;      // 0,16
    const int brow = t >> 2;            // 0..63
    const int bcol = (t & 3) * 8;       // 0,8,16,24

    const unsigned short* gA = xb  + (size_t)(blockM + arow) * K + acol;
    const unsigned short* gG = gwb + (size_t)(blockN + brow) * K + bcol;
    const unsigned short* gU = uwb + (size_t)(blockN + brow) * K + bcol;

    unsigned ldsA[2], ldsG[2], ldsU[2];
    #pragma unroll
    for (int b = 0; b < 2; ++b) {
        ldsA[b] = lds_off(&sA[b][arow * TK + acol]);
        ldsG[b] = lds_off(&sG[b][brow * TK + bcol]);
        ldsU[b] = lds_off(&sU[b][brow * TK + bcol]);
    }

    const int fr = lane & 15;           // fragment row
    const int kh = (lane >> 4) * 16;    // K half select
    const int nk = K / TK;

    // prologue: fill buffer 0
    async_ld_b128(ldsA[0],      gA);
    async_ld_b128(ldsA[0] + 16, gA + 8);
    async_ld_b128(ldsG[0],      gG);
    async_ld_b128(ldsU[0],      gU);

    for (int k0 = 0; k0 < nk; ++k0) {
        const int cur = k0 & 1;
        wait_async0();
        __syncthreads();
        if (k0 + 1 < nk) {                       // prefetch next tile (DMA)
            const int nxt = cur ^ 1;
            const size_t ko = (size_t)(k0 + 1) * TK;
            async_ld_b128(ldsA[nxt],      gA + ko);
            async_ld_b128(ldsA[nxt] + 16, gA + ko + 8);
            async_ld_b128(ldsG[nxt],      gG + ko);
            async_ld_b128(ldsU[nxt],      gU + ko);
        }
        v16bf af[2], gf[2], uf[2];
        #pragma unroll
        for (int mt = 0; mt < 2; ++mt)
            af[mt] = *(const v16bf*)(const void*)&sA[cur][(waveM + mt * 16 + fr) * TK + kh];
        #pragma unroll
        for (int nt = 0; nt < 2; ++nt) {
            gf[nt] = *(const v16bf*)(const void*)&sG[cur][(waveN + nt * 16 + fr) * TK + kh];
            uf[nt] = *(const v16bf*)(const void*)&sU[cur][(waveN + nt * 16 + fr) * TK + kh];
        }
        #pragma unroll
        for (int mt = 0; mt < 2; ++mt)
            #pragma unroll
            for (int nt = 0; nt < 2; ++nt) {
                accG[mt][nt] = __builtin_amdgcn_wmma_f32_16x16x32_bf16(
                    false, af[mt], false, gf[nt], (short)0, accG[mt][nt], false, false);
                accU[mt][nt] = __builtin_amdgcn_wmma_f32_16x16x32_bf16(
                    false, af[mt], false, uf[nt], (short)0, accU[mt][nt], false, false);
            }
    }

    // epilogue: rank-16 LoRA add (scalar; ~0.4% of WMMA work), SiLU * up
    const int mo = (lane >> 4) * 8;     // C layout: VGPR i -> M=i (lanes 0-15), M=8+i (16-31)
    #pragma unroll
    for (int mt = 0; mt < 2; ++mt) {
        #pragma unroll
        for (int i = 0; i < 8; ++i) {
            const int m = blockM + waveM + mt * 16 + mo + i;
            float xg[16], xu[16];
            #pragma unroll
            for (int r = 0; r < 16; ++r) { xg[r] = xag[m * 16 + r]; xu[r] = xau[m * 16 + r]; }
            #pragma unroll
            for (int nt = 0; nt < 2; ++nt) {
                const int n = blockN + waveN + nt * 16 + fr;
                float g = accG[mt][nt][i];
                float u = accU[mt][nt][i];
                #pragma unroll
                for (int r = 0; r < 16; ++r) {
                    g += xg[r] * gb[n * 16 + r];
                    u += xu[r] * ub[n * 16 + r];
                }
                const float s = g / (1.0f + __expf(-g));
                hidden[(size_t)m * H + n] = f2bf(s * u);
            }
        }
    }
}

// ---------------------------------------------------------------------------
// Down projection: out = hidden(bf16)[M,K] @ dwb(bf16)[N,K]^T + lora, fp32.
// ---------------------------------------------------------------------------
__global__ __launch_bounds__(256)
void down_wmma_kernel(const unsigned short* __restrict__ hidden,
                      const unsigned short* __restrict__ dwb,
                      const float* __restrict__ db,
                      const float* __restrict__ xad,
                      float* __restrict__ out,
                      int N, int K) {
    __shared__ __align__(32) unsigned short sA[2][TM * TK];
    __shared__ __align__(32) unsigned short sB[2][TN * TK];

    const int t     = threadIdx.x;
    const int lane  = t & 31;
    const int wave  = t >> 5;
    const int waveM = (wave >> 1) * 32;
    const int waveN = (wave & 1) * 32;
    const int blockM = blockIdx.y * TM;
    const int blockN = blockIdx.x * TN;

    v8f acc[2][2];
    #pragma unroll
    for (int i = 0; i < 2; ++i)
        #pragma unroll
        for (int j = 0; j < 2; ++j)
            #pragma unroll
            for (int e = 0; e < 8; ++e) acc[i][j][e] = 0.f;

    const int arow = t >> 1;
    const int acol = (t & 1) * 16;
    const int brow = t >> 2;
    const int bcol = (t & 3) * 8;

    const unsigned short* gA = hidden + (size_t)(blockM + arow) * K + acol;
    const unsigned short* gB = dwb    + (size_t)(blockN + brow) * K + bcol;

    unsigned ldsA[2], ldsB[2];
    #pragma unroll
    for (int b = 0; b < 2; ++b) {
        ldsA[b] = lds_off(&sA[b][arow * TK + acol]);
        ldsB[b] = lds_off(&sB[b][brow * TK + bcol]);
    }

    const int fr = lane & 15;
    const int kh = (lane >> 4) * 16;
    const int nk = K / TK;

    async_ld_b128(ldsA[0],      gA);
    async_ld_b128(ldsA[0] + 16, gA + 8);
    async_ld_b128(ldsB[0],      gB);

    for (int k0 = 0; k0 < nk; ++k0) {
        const int cur = k0 & 1;
        wait_async0();
        __syncthreads();
        if (k0 + 1 < nk) {
            const int nxt = cur ^ 1;
            const size_t ko = (size_t)(k0 + 1) * TK;
            async_ld_b128(ldsA[nxt],      gA + ko);
            async_ld_b128(ldsA[nxt] + 16, gA + ko + 8);
            async_ld_b128(ldsB[nxt],      gB + ko);
        }
        v16bf af[2], bfm[2];
        #pragma unroll
        for (int mt = 0; mt < 2; ++mt)
            af[mt] = *(const v16bf*)(const void*)&sA[cur][(waveM + mt * 16 + fr) * TK + kh];
        #pragma unroll
        for (int nt = 0; nt < 2; ++nt)
            bfm[nt] = *(const v16bf*)(const void*)&sB[cur][(waveN + nt * 16 + fr) * TK + kh];
        #pragma unroll
        for (int mt = 0; mt < 2; ++mt)
            #pragma unroll
            for (int nt = 0; nt < 2; ++nt)
                acc[mt][nt] = __builtin_amdgcn_wmma_f32_16x16x32_bf16(
                    false, af[mt], false, bfm[nt], (short)0, acc[mt][nt], false, false);
    }

    const int mo = (lane >> 4) * 8;
    #pragma unroll
    for (int mt = 0; mt < 2; ++mt) {
        #pragma unroll
        for (int i = 0; i < 8; ++i) {
            const int m = blockM + waveM + mt * 16 + mo + i;
            float xr[16];
            #pragma unroll
            for (int r = 0; r < 16; ++r) xr[r] = xad[m * 16 + r];
            #pragma unroll
            for (int nt = 0; nt < 2; ++nt) {
                const int n = blockN + waveN + nt * 16 + fr;
                float v = acc[mt][nt][i];
                #pragma unroll
                for (int r = 0; r < 16; ++r) v += xr[r] * db[n * 16 + r];
                out[(size_t)m * N + n] = v;
            }
        }
    }
}

// ---------------------------------------------------------------------------
extern "C" void kernel_launch(void* const* d_in, const int* in_sizes, int n_in,
                              void* d_out, int out_size, void* d_ws, size_t ws_size,
                              hipStream_t stream) {
    const float* x      = (const float*)d_in[0];
    const float* gate_w = (const float*)d_in[1];
    const float* up_w   = (const float*)d_in[2];
    const float* down_w = (const float*)d_in[3];
    const float* gate_a = (const float*)d_in[4];
    const float* gate_b = (const float*)d_in[5];
    const float* up_a   = (const float*)d_in[6];
    const float* up_b   = (const float*)d_in[7];
    const float* down_a = (const float*)d_in[8];
    const float* down_b = (const float*)d_in[9];
    float* out = (float*)d_out;

    const int D = 4096;
    const int H = 11008;
    const int M = in_sizes[0] / D;   // B*S = 4096

    // workspace layout (bf16 tensors + tiny fp32 LoRA intermediates)
    char* ws = (char*)d_ws;
    size_t off = 0;
    unsigned short* hidden = (unsigned short*)(ws + off); off += (size_t)M * H * 2;
    unsigned short* xb     = (unsigned short*)(ws + off); off += (size_t)M * D * 2;
    unsigned short* gwb    = (unsigned short*)(ws + off); off += (size_t)H * D * 2;
    unsigned short* uwb    = (unsigned short*)(ws + off); off += (size_t)H * D * 2;
    unsigned short* dwb    = (unsigned short*)(ws + off); off += (size_t)D * H * 2;
    float* xa_g = (float*)(ws + off); off += (size_t)M * 16 * 4;
    float* xa_u = (float*)(ws + off); off += (size_t)M * 16 * 4;
    float* xa_d = (float*)(ws + off);

    // 0) one-shot fp32 -> bf16 conversion (amortized over all tile reuse)
    {
        long long nx = (long long)M * D;
        long long nw = (long long)H * D;
        cvt_f32_bf16_kernel<<<(unsigned)(nx / 2048), 256, 0, stream>>>(x, xb, nx);
        cvt_f32_bf16_kernel<<<(unsigned)(nw / 2048), 256, 0, stream>>>(gate_w, gwb, nw);
        cvt_f32_bf16_kernel<<<(unsigned)(nw / 2048), 256, 0, stream>>>(up_w, uwb, nw);
        cvt_f32_bf16_kernel<<<(unsigned)(nw / 2048), 256, 0, stream>>>(down_w, dwb, nw);
    }

    // 1) LoRA A-projections of x (fp32 path, rank 16)
    xa_pair_f32_kernel<<<M, 256, 0, stream>>>(x, gate_a, up_a, xa_g, xa_u, D);

    // 2) fused gate/up GEMM + LoRA + SwiGLU -> hidden (bf16)
    gateup_wmma_kernel<<<dim3(H / TN, M / TM), 256, 0, stream>>>(
        xb, gwb, uwb, gate_b, up_b, xa_g, xa_u, hidden, H, D);

    // 3) LoRA A-projection of hidden
    xa_single_bf16_kernel<<<M, 256, 0, stream>>>(hidden, down_a, xa_d, H);

    // 4) down GEMM + LoRA -> out (fp32)
    down_wmma_kernel<<<dim3(D / TN, M / TM), 256, 0, stream>>>(
        hidden, dwb, down_b, xa_d, out, D, H);
}
